// DSFMarginal_79310866088089
// MI455X (gfx1250) — compile-verified, matrix operand order checked by reading, not run on hardware
//
#include <hip/hip_runtime.h>
#include <hip/hip_bf16.h>

typedef _Float16 half_t;
typedef __attribute__((ext_vector_type(16))) _Float16 v16h;
typedef __attribute__((ext_vector_type(8)))  float    v8f;

#define H_UNITS 48
#define N_LAYERS 6
#define EPSF 1e-06f
#define ROWS_PER_WG 128
#define THREADS 256

// ---------------------------------------------------------------------------
// Fragment loader for V_WMMA_F32_16X16X32_F16.
// 16-bit A-matrix 16x32 layout (ISA 7.12.2): lane L holds row M = L%16;
//   VGPR j (0..7): K = (j/4)*16 + (L/16)*8 + (j%4)*2 + {0,1}  (pairs contiguous)
// B fragments use the same addressing on an N-major (transposed) weight array,
// so one loader serves both LDS A-tiles and global B-tiles.
// ---------------------------------------------------------------------------
template <typename P>
__device__ inline v16h load_frag(const P* __restrict__ base, int ld, int rowBase, int kBase) {
  const int lane = threadIdx.x & 31;
  const int r  = rowBase + (lane & 15);
  const int hs = (lane >> 4) << 3;  // 0 or 8
  union { v16h v; unsigned int u[8]; } f;
  const P* p = base + (size_t)r * ld + kBase;
#pragma unroll
  for (int j = 0; j < 8; ++j) {
    const int k = ((j >> 2) << 4) + hs + ((j & 3) << 1);
    unsigned int tmp;
    __builtin_memcpy(&tmp, (const void*)(p + k), 4);
    f.u[j] = tmp;
  }
  return f.v;
}

// ---------------------------------------------------------------------------
// One wave computes a 16 x (NCT*16) output block:
//   acc[NCT] registers; the A-fragment is loaded once per k-step and reused
//   across all NCT col-tiles (8-9 WMMAs per pair of ds_load_b128).
// C/D layout (ISA 7.12.2): lane L covers col N = L%16; VGPR r is row
//   M = r + 8*(L/16).
// ---------------------------------------------------------------------------
template <int NCT, bool F16OUT>
__device__ inline void gemm_block(const half_t* __restrict__ A, int lda,
                                  const half_t* __restrict__ Bt, int ldb, int K,
                                  int rowTile16, int outColBase, int bColBase,
                                  const float* __restrict__ bias, int biasBase,
                                  half_t* outH, int ldoH,
                                  float*  outF, int ldoF) {
  v8f acc[NCT];
#pragma unroll
  for (int j = 0; j < NCT; ++j) acc[j] = v8f{0.f,0.f,0.f,0.f,0.f,0.f,0.f,0.f};

#pragma unroll 2
  for (int k = 0; k < K; k += 32) {
    v16h a = load_frag(A, lda, rowTile16, k);
#pragma unroll
    for (int j = 0; j < NCT; ++j) {
      if (j == 0) {
        // stream-ahead hint on the weight rows (global_prefetch_b8)
        __builtin_prefetch((const void*)(Bt + (size_t)(bColBase + (threadIdx.x & 15)) * ldb + k + 64), 0, 1);
      }
      v16h b = load_frag(Bt, ldb, bColBase + j * 16, k);
      acc[j] = __builtin_amdgcn_wmma_f32_16x16x32_f16(
          /*neg_a=*/false, a, /*neg_b=*/false, b,
          /*c_mod=*/(short)0, acc[j], /*reuse_a=*/false, /*reuse_b=*/false);
    }
  }

  const int lane = threadIdx.x & 31;
  const int nloc = lane & 15;
  const int m0   = rowTile16 + ((lane >> 4) << 3);
#pragma unroll
  for (int j = 0; j < NCT; ++j) {
    const float bv = bias[biasBase + j * 16 + nloc];
#pragma unroll
    for (int r = 0; r < 8; ++r) {
      float v = acc[j][r] + bv;
      if (F16OUT) {
        v = fmaxf(v, 0.f);  // relu
        outH[(size_t)(m0 + r) * ldoH + outColBase + j * 16 + nloc] = (half_t)v;
      } else {
        outF[(size_t)(m0 + r) * ldoF + outColBase + j * 16 + nloc] = v;
      }
    }
  }
}

// ---------------------------------------------------------------------------
// math helpers (branchless, stable)
// ---------------------------------------------------------------------------
__device__ inline float softplusf(float v) {           // log(1+e^v)
  return fmaxf(v, 0.f) + log1pf(__expf(-fabsf(v)));
}
__device__ inline float logsigf(float z) {             // log sigmoid(z)
  return fminf(z, 0.f) - log1pf(__expf(-fabsf(z)));
}
__device__ inline float predmax(float v) {             // reduce over lane pair
  return fmaxf(v, __shfl_xor(v, 1, 32));
}
__device__ inline float predsum(float v) {
  return v + __shfl_xor(v, 1, 32);
}

// ---------------------------------------------------------------------------
// weight transpose + f32->f16 convert: in[K][N] -> out[N][K]
// ---------------------------------------------------------------------------
__global__ void convert_transpose(const float* __restrict__ in, half_t* __restrict__ out,
                                  int K, int N) {
  const size_t total = (size_t)K * N;
  for (size_t i = (size_t)blockIdx.x * blockDim.x + threadIdx.x; i < total;
       i += (size_t)gridDim.x * blockDim.x) {
    const size_t n = i / (size_t)K;
    const size_t k = i - n * (size_t)K;
    out[i] = (half_t)in[k * (size_t)N + n];  // coalesced writes
  }
}

// ---------------------------------------------------------------------------
// Fused MLP (3 WMMA GEMMs) + 6 sigmoid-flow layers. One WG = 128 rows.
// Dynamic LDS layout (bytes) -- buffers with disjoint lifetimes alias:
//   s_h1  : [0,      131072)  128x512 f16   (live: GEMM1 out -> GEMM2 in)
//   s_p   : [0,       73728)  128x144 f32   (live: GEMM3 out -> flow; h1 dead)
//   s_h2  : [131072, 262144)  128x512 f16   (live: GEMM2 out -> GEMM3 in)
//   s_ctx : [131072, 196608)  128x256 f16   (live: stage -> GEMM1; h2 not yet)
// Peak 256 KB of the WGP's 320 KB; one WG (8 waves) per WGP.
// ---------------------------------------------------------------------------
__global__ __launch_bounds__(THREADS)
void dsf_fused(const float* __restrict__ ctx, const float* __restrict__ xin,
               const float* __restrict__ b1, const float* __restrict__ b2,
               const float* __restrict__ b3,
               const half_t* __restrict__ w1t,   // 512 x 256
               const half_t* __restrict__ w2t,   // 512 x 512
               const half_t* __restrict__ w3t,   // 864 x 512
               float* __restrict__ out, int totalRows) {
  extern __shared__ __align__(16) char smem[];
  half_t* s_h1  = (half_t*)(smem);
  float*  s_p   = (float*)(smem);
  half_t* s_h2  = (half_t*)(smem + 131072);
  half_t* s_ctx = (half_t*)(smem + 131072);

  const int rowBase = blockIdx.x * ROWS_PER_WG;
  const int tid  = threadIdx.x;
  const int wave = tid >> 5;
  const int rowTile = wave * 16;             // each wave owns one 16-row tile

  // ---- stage context tile into LDS as f16 (vectorized f32x4 loads) ----
  {
    const float4* cg = (const float4*)(ctx + (size_t)rowBase * 256);
#pragma unroll
    for (int it = 0; it < (ROWS_PER_WG * 64) / THREADS; ++it) {
      const int i = tid + it * THREADS;
      const float4 v = cg[i];
      const int m  = i >> 6;
      const int c4 = (i & 63) << 2;
      half_t* d = s_ctx + m * 256 + c4;
      d[0] = (half_t)v.x; d[1] = (half_t)v.y; d[2] = (half_t)v.z; d[3] = (half_t)v.w;
    }
  }
  __syncthreads();

  // ---- GEMM1: h1 = relu(ctx @ W1 + b1)   [128x256]x[256x512] ----
  // All 8 waves walk identical B addresses in lockstep -> WGP$ dedups traffic.
#pragma unroll
  for (int blk = 0; blk < 4; ++blk) {
    const int ct = blk * 128;                // 8 col-tiles per block
    gemm_block<8, true>(s_ctx, 256, w1t, 256, 256, rowTile, ct, ct,
                        b1, ct, s_h1, 512, nullptr, 0);
  }
  __syncthreads();

  // ---- GEMM2: h2 = relu(h1 @ W2 + b2)   [128x512]x[512x512] ----
#pragma unroll
  for (int blk = 0; blk < 4; ++blk) {
    const int ct = blk * 128;
    gemm_block<8, true>(s_h1, 512, w2t, 512, 512, rowTile, ct, ct,
                        b2, ct, s_h2, 512, nullptr, 0);
  }

  // ---- flow-layer state: 2 lanes per row, 24 units each ----
  const int row  = tid >> 1;                 // 0..127
  const int sub  = tid & 1;
  const int rowG = rowBase + row;
  float x    = xin[rowG];
  float ldet = 0.f;

  for (int L = 0; L < N_LAYERS; ++L) {
    __syncthreads();  // h2 ready / previous flow reads of s_p done

    // ---- GEMM3 chunk: params[:, L*144 : L*144+144] -> s_p (f32) ----
    // 9 col-tiles per wave, one accumulator block (A-frag reused 9x).
    gemm_block<9, false>(s_h2, 512, w3t, 512, 512, rowTile, 0,
                         L * 144, b3, L * 144, nullptr, 0, s_p, 144);
    __syncthreads();

    // ---- sigmoid flow layer (48 units; 24 per lane, pair reductions) ----
    const float* pr = s_p + row * 144;
    float aj[24], prej[24], wj[24];
    float wmax = -1e30f;
#pragma unroll
    for (int j = 0; j < 24; ++j) {
      const int idx = sub * 24 + j;
      const float pa = pr[idx];
      const float pb = pr[H_UNITS + idx];
      const float pw = pr[2 * H_UNITS + idx];
      const float a  = softplusf(pa);
      aj[j] = a; wj[j] = pw; prej[j] = a * x + pb;
      wmax = fmaxf(wmax, pw);
    }
    wmax = predmax(wmax);

    float sw = 0.f, snum = 0.f;
#pragma unroll
    for (int j = 0; j < 24; ++j) {
      const float e   = __expf(wj[j] - wmax);
      const float sig = 1.f / (1.f + __expf(-prej[j]));
      sw += e; snum += e * sig;
    }
    sw = predsum(sw); snum = predsum(snum);
    const float xpre  = snum / sw;
    const float logsw = __logf(sw);

    float lj[24];
    float lm = -1e30f;
#pragma unroll
    for (int j = 0; j < 24; ++j) {
      const float v = (wj[j] - wmax - logsw) + logsigf(prej[j]) + logsigf(-prej[j]) + __logf(aj[j]);
      lj[j] = v; lm = fmaxf(lm, v);
    }
    lm = predmax(lm);
    float ls = 0.f;
#pragma unroll
    for (int j = 0; j < 24; ++j) ls += __expf(lj[j] - lm);
    ls = predsum(ls);
    ldet += lm + __logf(ls);

    if (L == N_LAYERS - 1) {
      x = xpre;
    } else {
      const float xc = xpre * (1.f - EPSF) + EPSF * 0.5f;
      const float l1 = __logf(xc);
      const float l2 = __logf(1.f - xc);
      x = l1 - l2;
      ldet += (-9.999995e-07f) - l1 - l2;   // log(1-EPS) - log(xc) - log(1-xc)
    }
  }

  if (sub == 0 && rowG < totalRows) {
    out[rowG]             = x;
    out[totalRows + rowG] = ldet;
  }
}

extern "C" void kernel_launch(void* const* d_in, const int* in_sizes, int n_in,
                              void* d_out, int out_size, void* d_ws, size_t ws_size,
                              hipStream_t stream) {
  const float* context = (const float*)d_in[0];
  const float* x       = (const float*)d_in[1];
  const float* W1      = (const float*)d_in[2];
  const float* b1      = (const float*)d_in[3];
  const float* W2      = (const float*)d_in[4];
  const float* b2      = (const float*)d_in[5];
  const float* W3      = (const float*)d_in[6];
  const float* b3      = (const float*)d_in[7];
  float* out = (float*)d_out;

  const int totalRows = in_sizes[1];        // B*N = 131072

  // workspace: f16 transposed weights (N-major so B-fragments load like A)
  half_t* w1t = (half_t*)d_ws;                       // 512*256
  half_t* w2t = w1t + 512 * 256;                     // 512*512
  half_t* w3t = w2t + 512 * 512;                     // 864*512

  convert_transpose<<<512, THREADS, 0, stream>>>(W1, w1t, 256, 512);
  convert_transpose<<<512, THREADS, 0, stream>>>(W2, w2t, 512, 512);
  convert_transpose<<<1024, THREADS, 0, stream>>>(W3, w3t, 512, 864);

  const int nWG = totalRows / ROWS_PER_WG;           // 1024
  const size_t lds = 262144;                         // 256 KB dynamic LDS
  dsf_fused<<<nWG, THREADS, lds, stream>>>(context, x, b1, b2, b3,
                                           w1t, w2t, w3t, out, totalRows);
}
